// CrossAttention_15642270892415
// MI455X (gfx1250) — matrix operand order repaired and checked by reading, not compiled, not run on hardware
//
#include <hip/hip_runtime.h>

#define BSZ 32
#define DIM 1024
#define SEQ 1024
#define KP  40   // padded K stride in halves (80 bytes, 16B-aligned rows)

typedef __attribute__((ext_vector_type(16))) _Float16 v16h;
typedef __attribute__((ext_vector_type(8)))  float    v8f;

__device__ __forceinline__ unsigned short f2h(float x) {
  _Float16 h = (_Float16)x;
  return __builtin_bit_cast(unsigned short, h);
}
// packed f32x2 -> f16x2 (v_cvt_pk_rtz_f16_f32), one VALU op per pair
__device__ __forceinline__ unsigned pk2h(float a, float b) {
  auto p = __builtin_amdgcn_cvt_pkrtz(a, b);   // __fp16 ext_vector_type(2)
  return __builtin_bit_cast(unsigned, p);
}

// ---- WMMA fragment loads (wave32, 16x16x32 f16 layouts per CDNA5 ISA 7.12.2) ----
// A (16x32, MxK): lane L holds row M=L%16; VGPRs 0-3 = K kb..kb+7, VGPRs 4-7 = K 16+kb..16+kb+7, kb=(L<16)?0:8
__device__ __forceinline__ v16h ldsA_frag(const unsigned short* lds, int mBase, int lane) {
  int row = mBase + (lane & 15);
  int kb  = (lane < 16) ? 0 : 8;
  const unsigned short* p = lds + row * KP + kb;
  union { uint4 q[2]; v16h h; } f;
  f.q[0] = *(const uint4*)(p);        // K = kb .. kb+7
  f.q[1] = *(const uint4*)(p + 16);   // K = 16+kb .. 16+kb+7
  return f.h;
}
// B (32x16, KxN): lane L holds column N=L%16; VGPRs 0-3 = K kb..kb+7, VGPRs 4-7 = K 16+kb..16+kb+7, kb=(L<16)?0:8
// LDS B tile stored [N][K] so each lane reads two runs of 8 contiguous halves (K kb.. and 16+kb..).
__device__ __forceinline__ v16h ldsB_frag(const unsigned short* lds, int nBase, int lane) {
  int col = nBase + (lane & 15);
  int kb  = (lane < 16) ? 0 : 8;
  const unsigned short* p = lds + col * KP + kb;
  union { uint4 q[2]; v16h h; } f;
  f.q[0] = *(const uint4*)(p);        // K = kb .. kb+7
  f.q[1] = *(const uint4*)(p + 16);   // K = 16+kb .. 16+kb+7
  return f.h;
}

__device__ __forceinline__ v8f wmma32(v16h a, v16h b, v8f c) {
  return __builtin_amdgcn_wmma_f32_16x16x32_f16(false, a, false, b, (short)0, c, false, false);
}

// ---- LDS staging helpers (256-thread blocks) ----
// 128 rows x 32 cols fp32 -> lds[row*KP + col] as f16 ([M][K]; also [N][K] when rows are N).
// Packed converts + single ds_store_b64 per 4 elements.
__device__ __forceinline__ void stage_f32_MK(const float* src, int stride, unsigned short* lds, int tid) {
  #pragma unroll
  for (int it = 0; it < 4; ++it) {
    int q = tid + it * 256;
    int row = q >> 3;
    int c4  = (q & 7) << 2;
    float4 v = *(const float4*)(src + (size_t)row * stride + c4);
    uint2 p;
    p.x = pk2h(v.x, v.y);
    p.y = pk2h(v.z, v.w);
    *(uint2*)(lds + row * KP + c4) = p;
  }
}
// 32 rows(K) x 128 cols(N) fp32 -> lds[col*KP + row] (transpose to [N][K]) as f16
__device__ __forceinline__ void stage_f32_KN_T(const float* src, int stride, unsigned short* lds, int tid) {
  #pragma unroll
  for (int it = 0; it < 4; ++it) {
    int q = tid + it * 256;
    int row = q >> 5;
    int c4  = (q & 31) << 2;
    float4 v = *(const float4*)(src + (size_t)row * stride + c4);
    lds[(c4 + 0) * KP + row] = f2h(v.x);
    lds[(c4 + 1) * KP + row] = f2h(v.y);
    lds[(c4 + 2) * KP + row] = f2h(v.z);
    lds[(c4 + 3) * KP + row] = f2h(v.w);
  }
}

// Async copy of a 128-row x 32-half tile (row stride DIM halves) into LDS rows of KP halves.
// Each wave-level GLOBAL_LOAD_ASYNC_TO_LDS_B128 moves 8 rows (32 lanes x 16B); 2 issues per wave,
// 8 waves cover all 128 rows. Tracked on ASYNCcnt (memory -> LDS, no VGPR round trip).
__device__ __forceinline__ void async_stage_tile(const unsigned short* __restrict__ src,
                                                 unsigned ldsBase, int tid) {
  int lane = tid & 31, w = tid >> 5;
  int c = lane & 3;                       // 16-byte chunk within the 64B row
  #pragma unroll
  for (int t = 0; t < 2; ++t) {
    int row = w * 16 + t * 8 + (lane >> 2);
    unsigned long long ga = (unsigned long long)(const void*)(src + (size_t)row * DIM + c * 8);
    unsigned la = ldsBase + (unsigned)(row * (KP * 2) + c * 16);
    asm volatile("global_load_async_to_lds_b128 %0, %1, off"
                 :: "v"(la), "v"(ga) : "memory");
  }
}

// ===== K1: Ylin[b][s][o] = f16(relu(sum_d W[o,d]*X[b,d,s] + bias[o]))  (stored s-major!) =====
__global__ __launch_bounds__(256) void k_linear_relu(const float* __restrict__ X,
                                                     const float* __restrict__ W,
                                                     const float* __restrict__ bias,
                                                     unsigned short* __restrict__ Y) {
  __shared__ __align__(16) unsigned short lds_a[128 * KP];
  __shared__ __align__(16) unsigned short lds_b[128 * KP];
  int tid = threadIdx.x, lane = tid & 31, wid = tid >> 5;
  int wm = (wid & 1) * 64, wn = (wid >> 1) * 32;
  int m0 = blockIdx.y * 128, n0 = blockIdx.x * 128;   // m = o, n = s
  size_t boff = (size_t)blockIdx.z << 20;
  const float* Xb = X + boff;
  v8f acc[4][2] = {};
  for (int k0 = 0; k0 < DIM; k0 += 32) {
    stage_f32_MK  (W  + (size_t)m0 * DIM + k0, DIM, lds_a, tid);
    stage_f32_KN_T(Xb + (size_t)k0 * SEQ + n0, SEQ, lds_b, tid);
    __syncthreads();
    v16h bf0 = ldsB_frag(lds_b, wn, lane);
    v16h bf1 = ldsB_frag(lds_b, wn + 16, lane);
    #pragma unroll
    for (int mt = 0; mt < 4; ++mt) {
      v16h af = ldsA_frag(lds_a, wm + mt * 16, lane);
      acc[mt][0] = wmma32(af, bf0, acc[mt][0]);
      acc[mt][1] = wmma32(af, bf1, acc[mt][1]);
    }
    __syncthreads();
  }
  // C layout: VGPR r -> (M = base + lh + r, N = lane%16): 8 contiguous o-values per lane.
  // With Y stored [s][o], each lane packs its 8 f16 results into one b128 store.
  int lh = (lane >> 4) * 8, nlo = lane & 15;
  unsigned short* Yb = Y + boff;
  #pragma unroll
  for (int mt = 0; mt < 4; ++mt) {
    int mb = m0 + wm + mt * 16 + lh;
    #pragma unroll
    for (int nt = 0; nt < 2; ++nt) {
      int n = n0 + wn + nt * 16 + nlo;
      union { unsigned u[4]; uint4 q; } pk;
      #pragma unroll
      for (int r2 = 0; r2 < 4; ++r2) {
        float v0 = fmaxf(acc[mt][nt][2 * r2]     + bias[mb + 2 * r2],     0.0f);
        float v1 = fmaxf(acc[mt][nt][2 * r2 + 1] + bias[mb + 2 * r2 + 1], 0.0f);
        pk.u[r2] = pk2h(v0, v1);
      }
      *(uint4*)(Yb + (size_t)n * DIM + mb) = pk.q;
    }
  }
}

// ===== K2: Aff[b][i][j] = (1/32) * sum_o Rlin[b][i][o] * Plin[b][j][o] =====
// s-major f16 operands: tiles are pure row copies -> staged with GLOBAL_LOAD_ASYNC_TO_LDS_B128.
__global__ __launch_bounds__(256) void k_affinity(const unsigned short* __restrict__ Rlin,
                                                  const unsigned short* __restrict__ Plin,
                                                  float* __restrict__ Aff) {
  __shared__ __align__(16) unsigned short lds_a[128 * KP];
  __shared__ __align__(16) unsigned short lds_b[128 * KP];
  int tid = threadIdx.x, lane = tid & 31, wid = tid >> 5;
  int wm = (wid & 1) * 64, wn = (wid >> 1) * 32;
  int m0 = blockIdx.y * 128, n0 = blockIdx.x * 128;   // m = i (review), n = j (paper)
  size_t boff = (size_t)blockIdx.z << 20;
  const unsigned short* Rb = Rlin + boff;
  const unsigned short* Pb = Plin + boff;
  unsigned ldsA = (unsigned)(unsigned long long)(void*)lds_a;
  unsigned ldsB = (unsigned)(unsigned long long)(void*)lds_b;
  v8f acc[4][2] = {};
  for (int k0 = 0; k0 < DIM; k0 += 32) {              // k = o
    async_stage_tile(Rb + (size_t)m0 * DIM + k0, ldsA, tid);  // [i][o] -> [M][K]
    async_stage_tile(Pb + (size_t)n0 * DIM + k0, ldsB, tid);  // [j][o] -> [N][K]
    asm volatile("s_wait_asynccnt 0x0" ::: "memory");
    __syncthreads();
    v16h bf0 = ldsB_frag(lds_b, wn, lane);
    v16h bf1 = ldsB_frag(lds_b, wn + 16, lane);
    #pragma unroll
    for (int mt = 0; mt < 4; ++mt) {
      v16h af = ldsA_frag(lds_a, wm + mt * 16, lane);
      acc[mt][0] = wmma32(af, bf0, acc[mt][0]);
      acc[mt][1] = wmma32(af, bf1, acc[mt][1]);
    }
    __syncthreads();
  }
  int lh = (lane >> 4) * 8, nlo = lane & 15;
  float* Ab = Aff + boff;
  #pragma unroll
  for (int mt = 0; mt < 4; ++mt)
    #pragma unroll
    for (int nt = 0; nt < 2; ++nt) {
      int n = n0 + wn + nt * 16 + nlo;
      #pragma unroll
      for (int r = 0; r < 8; ++r) {
        int m = m0 + wm + mt * 16 + lh + r;
        Ab[(size_t)m * SEQ + n] = acc[mt][nt][r] * 0.03125f;  // / sqrt(1024)
      }
    }
}

// ===== K3: in-place column softmax over i (axis=1) for each (b, j) =====
__global__ void k_softmax_col(float* __restrict__ Aff) {
  int tx = threadIdx.x, ty = threadIdx.y;      // block (32,8)
  int b = blockIdx.z;
  int j = blockIdx.x * 32 + tx;
  float* A = Aff + ((size_t)b << 20) + j;
  __shared__ float redm[8][32];
  __shared__ float reds[8][32];
  float m = -3.0e38f;
  for (int i = ty; i < SEQ; i += 8) m = fmaxf(m, A[(size_t)i * SEQ]);
  redm[ty][tx] = m;
  __syncthreads();
  if (ty == 0) {
    #pragma unroll
    for (int r = 1; r < 8; ++r) m = fmaxf(m, redm[r][tx]);
    redm[0][tx] = m;
  }
  __syncthreads();
  m = redm[0][tx];
  float s = 0.0f;
  for (int i = ty; i < SEQ; i += 8) s += __expf(A[(size_t)i * SEQ] - m);
  reds[ty][tx] = s;
  __syncthreads();
  if (ty == 0) {
    #pragma unroll
    for (int r = 1; r < 8; ++r) s += reds[r][tx];
    reds[0][tx] = s;
  }
  __syncthreads();
  float inv = 1.0f / reds[0][tx];
  for (int i = ty; i < SEQ; i += 8) {
    size_t o = (size_t)i * SEQ;
    A[o] = __expf(A[o] - m) * inv;
  }
}

// ===== K4: row sums of C: sumC[b][i] = sum_j C[b][i][j] (one wave per row) =====
__global__ __launch_bounds__(256) void k_rowsum(const float* __restrict__ C,
                                                float* __restrict__ sumC) {
  int lane = threadIdx.x & 31, wid = threadIdx.x >> 5;
  int b = blockIdx.y;
  int i = blockIdx.x * 8 + wid;
  const float* row = C + ((size_t)b << 20) + (size_t)i * SEQ;
  float s = 0.0f;
  for (int j = lane; j < SEQ; j += 32) s += row[j];
  #pragma unroll
  for (int off = 16; off > 0; off >>= 1) s += __shfl_down(s, off, 32);
  if (lane == 0) sumC[b * SEQ + i] = s;
}

// ===== K5: Pr[b][i][d] = sum_j C[b][i][j] * paper[b][d][j]; also fills Rc rows 2048..3071 =====
__global__ __launch_bounds__(256) void k_pr(const float* __restrict__ C,
                                            const float* __restrict__ P,
                                            float* __restrict__ outPr,
                                            float* __restrict__ outRc) {
  __shared__ __align__(16) unsigned short lds_a[128 * KP];
  __shared__ __align__(16) unsigned short lds_b[128 * KP];
  int tid = threadIdx.x, lane = tid & 31, wid = tid >> 5;
  int wm = (wid & 1) * 64, wn = (wid >> 1) * 32;
  int m0 = blockIdx.y * 128, n0 = blockIdx.x * 128;   // m = i, n = d
  size_t boff = (size_t)blockIdx.z << 20;
  const float* Cb = C + boff;
  const float* Pb = P + boff;
  v8f acc[4][2] = {};
  for (int k0 = 0; k0 < SEQ; k0 += 32) {              // k = j
    stage_f32_MK(Cb + (size_t)m0 * SEQ + k0, SEQ, lds_a, tid);   // [i][j] -> [M][K]
    stage_f32_MK(Pb + (size_t)n0 * SEQ + k0, SEQ, lds_b, tid);   // paper[d][j] -> [N][K]
    __syncthreads();
    v16h bf0 = ldsB_frag(lds_b, wn, lane);
    v16h bf1 = ldsB_frag(lds_b, wn + 16, lane);
    #pragma unroll
    for (int mt = 0; mt < 4; ++mt) {
      v16h af = ldsA_frag(lds_a, wm + mt * 16, lane);
      acc[mt][0] = wmma32(af, bf0, acc[mt][0]);
      acc[mt][1] = wmma32(af, bf1, acc[mt][1]);
    }
    __syncthreads();
  }
  int lh = (lane >> 4) * 8, nlo = lane & 15;
  int b = blockIdx.z;
  #pragma unroll
  for (int mt = 0; mt < 4; ++mt)
    #pragma unroll
    for (int nt = 0; nt < 2; ++nt) {
      int n = n0 + wn + nt * 16 + nlo;
      #pragma unroll
      for (int r = 0; r < 8; ++r) {
        int m = m0 + wm + mt * 16 + lh + r;
        float v = acc[mt][nt][r];
        outPr[((size_t)b * SEQ + m) * DIM + n] = v;
        outRc[((size_t)b * 3 * SEQ + 2 * SEQ + m) * DIM + n] = v;
      }
    }
}

// ===== K6: transpose review -> r, Rp = r * sumC, and Rc rows 0..2047 =====
__global__ void k_rp(const float* __restrict__ review, const float* __restrict__ sumC,
                     float* __restrict__ outRp, float* __restrict__ outRc) {
  __shared__ float t[32][33];
  int tx = threadIdx.x, ty = threadIdx.y;   // block (32,8)
  int b = blockIdx.z;
  int i0 = blockIdx.x * 32, d0 = blockIdx.y * 32;
  const float* rv = review + ((size_t)b << 20);
  #pragma unroll
  for (int it = 0; it < 4; ++it) {
    int d = d0 + ty + it * 8;
    t[ty + it * 8][tx] = rv[(size_t)d * SEQ + i0 + tx];
  }
  __syncthreads();
  #pragma unroll
  for (int it = 0; it < 4; ++it) {
    int i = i0 + ty + it * 8;
    int d = d0 + tx;
    float v  = t[tx][ty + it * 8];
    float s  = sumC[b * SEQ + i];
    float rp = v * s;
    outRp[((size_t)b * SEQ + i) * DIM + d]           = rp;
    outRc[((size_t)b * 3 * SEQ + i) * DIM + d]       = v;
    outRc[((size_t)b * 3 * SEQ + SEQ + i) * DIM + d] = rp;
  }
}

extern "C" void kernel_launch(void* const* d_in, const int* in_sizes, int n_in,
                              void* d_out, int out_size, void* d_ws, size_t ws_size,
                              hipStream_t stream) {
  const float* paper  = (const float*)d_in[0];
  const float* review = (const float*)d_in[1];
  const float* W      = (const float*)d_in[2];
  const float* bias   = (const float*)d_in[3];

  const size_t perB = (size_t)DIM * SEQ;                     // 1<<20
  unsigned short* Plin = (unsigned short*)d_ws;              // 64 MB f16 [b][s][o]
  unsigned short* Rlin = Plin + (size_t)BSZ * perB;          // 64 MB f16 [b][s][o]
  float* Aff  = (float*)(Rlin + (size_t)BSZ * perB);         // 128 MB f32 [b][i][j] -> C in place
  float* sumC = Aff + (size_t)BSZ * perB;                    // 128 KB f32 [b][i]

  float* outRp = (float*)d_out;                              // (b, sr, d)
  float* outPr = outRp + (size_t)BSZ * perB;                 // (b, sr, d)
  float* outRc = outPr + (size_t)BSZ * perB;                 // (b, 3*sr, d)

  dim3 gGemm(SEQ / 128, SEQ / 128, BSZ);
  k_linear_relu<<<gGemm, 256, 0, stream>>>(paper,  W, bias, Plin);
  k_linear_relu<<<gGemm, 256, 0, stream>>>(review, W, bias, Rlin);
  k_affinity   <<<gGemm, 256, 0, stream>>>(Rlin, Plin, Aff);
  k_softmax_col<<<dim3(SEQ / 32, 1, BSZ), dim3(32, 8), 0, stream>>>(Aff);
  k_rowsum     <<<dim3(SEQ / 8, BSZ), 256, 0, stream>>>(Aff, sumC);
  k_pr         <<<gGemm, 256, 0, stream>>>(Aff, paper, outPr, outRc);
  k_rp         <<<dim3(SEQ / 32, DIM / 32, BSZ), dim3(32, 8), 0, stream>>>(review, sumC, outRp, outRc);
}
